// AttnBlock_35519379538166
// MI455X (gfx1250) — compile-verified, matrix operand order checked by reading, not run on hardware
//
#include <hip/hip_runtime.h>
#include <hip/hip_bf16.h>

typedef __bf16 bf16;
typedef __attribute__((ext_vector_type(16))) __bf16 v16bf;
typedef __attribute__((ext_vector_type(8)))  float  v8f;

#define BATCH 4
#define CCH   512
#define NPIX  4096   // 64*64

// ---------------------------------------------------------------------------
// Fragment helpers.
// CDNA5 16x32 bf16 A-fragment: lane L holds row M = L%16, K chunks
// {K0..K0+7, K0+16..K0+23} with K0 = 8*(L/16)  -> two 16B contiguous loads.
// B-fragment for D = A x B loaded from Bt[N][K] row-major uses the same
// per-lane pattern (N = L%16): "NT" GEMM  D[M][N] = A[M][K] * Bt[N][K]^T.
// ---------------------------------------------------------------------------
__device__ __forceinline__ v16bf load_frag(const bf16* __restrict__ p, int ld) {
    const int lane = threadIdx.x & 31;
    const int M  = lane & 15;
    const int K0 = (lane >> 4) << 3;
    const bf16* r = p + (size_t)M * ld + K0;
    union { v16bf v; uint4 u[2]; } f;
    f.u[0] = *reinterpret_cast<const uint4*>(r);
    f.u[1] = *reinterpret_cast<const uint4*>(r + 16);
    return f.v;
}

__device__ __forceinline__ v8f wmma_bf16(v16bf a, v16bf b, v8f c) {
    return __builtin_amdgcn_wmma_f32_16x16x32_bf16(
        /*neg_a=*/false, a, /*neg_b=*/false, b,
        /*c_mod=*/(short)0, c, /*reuse_a=*/false, /*reuse_b=*/false);
}

// ---------------------------------------------------------------------------
// fp32 -> bf16 convert (weights)
// ---------------------------------------------------------------------------
__global__ __launch_bounds__(256) void cvt_bf16_kernel(const float* __restrict__ s,
                                                       bf16* __restrict__ d, int n) {
    int i = blockIdx.x * 256 + threadIdx.x;
    if (i < n) d[i] = (bf16)s[i];
}

// ---------------------------------------------------------------------------
// GroupNorm: x[b][c][n] fp32 -> hn[b][n][c] bf16 (transpose fused)
// ---------------------------------------------------------------------------
__global__ __launch_bounds__(256) void gn_kernel(const float* __restrict__ x,
                                                 const float* __restrict__ gw,
                                                 const float* __restrict__ gb,
                                                 bf16* __restrict__ hn) {
    const int batch = blockIdx.x >> 5;
    const int g     = blockIdx.x & 31;
    const int c0    = g * 16;
    const int tid   = threadIdx.x;
    const float* xg = x + ((size_t)batch * CCH + c0) * NPIX;

    __shared__ float s1[256], s2[256];
    float sum = 0.f, ssq = 0.f;
    for (int e = tid; e < 16 * NPIX; e += 256) {
        float v = xg[e];
        sum += v; ssq += v * v;
    }
    s1[tid] = sum; s2[tid] = ssq;
    __syncthreads();
    for (int s = 128; s > 0; s >>= 1) {
        if (tid < s) { s1[tid] += s1[tid + s]; s2[tid] += s2[tid + s]; }
        __syncthreads();
    }
    const float inv  = 1.0f / (16.0f * NPIX);
    const float mean = s1[0] * inv;
    const float var  = s2[0] * inv - mean * mean;
    const float rstd = rsqrtf(var + 1e-5f);

    for (int e = tid; e < 16 * NPIX; e += 256) {
        int c = e >> 12;
        int n = e & (NPIX - 1);
        float y = (xg[e] - mean) * rstd * gw[c0 + c] + gb[c0 + c];
        hn[((size_t)batch * NPIX + n) * CCH + c0 + c] = (bf16)y;
    }
}

// ---------------------------------------------------------------------------
// NT GEMM, ping-pong pipelined (no buffer-rotation copies), K = 512 fixed:
// out[m][n] = sum_k A[m][k]*Bt[n][k] + bias
// Each wave computes a 32(M) x 64(N) block: 8 accumulator tiles.
// ---------------------------------------------------------------------------
__global__ __launch_bounds__(128) void gemm_nt_kernel(
        const bf16* __restrict__ A, long long strideA,
        const bf16* __restrict__ Bt, long long strideB,
        const float* __restrict__ bias, int bias_on_m,
        bf16* __restrict__ out, long long strideO,
        int M, int Nn, int ngroups) {
    const int batch = blockIdx.z;
    A   += (size_t)batch * strideA;
    Bt  += (size_t)batch * strideB;
    out += (size_t)batch * strideO;

    const int wave = threadIdx.x >> 5;
    const int lane = threadIdx.x & 31;
    const int wg   = blockIdx.x * 4 + wave;
    const int mt   = wg / ngroups;      // 32-row tile index
    const int ng   = wg % ngroups;      // 64-col group index
    if (mt * 32 >= M) return;

    const bf16* arow = A  + (size_t)(mt * 32) * CCH;
    const bf16* brow = Bt + (size_t)(ng * 64) * CCH;

    v16bf a0[2], b0[4], a1[2], b1[4];
#pragma unroll
    for (int i = 0; i < 2; ++i) a0[i] = load_frag(arow + (size_t)(i * 16) * CCH, CCH);
#pragma unroll
    for (int t = 0; t < 4; ++t) b0[t] = load_frag(brow + (size_t)(t * 16) * CCH, CCH);

    v8f acc[2][4] = {};
    for (int k0 = 0; k0 < CCH - 64; k0 += 64) {
        // load buf1 @ k0+32
#pragma unroll
        for (int i = 0; i < 2; ++i)
            a1[i] = load_frag(arow + (size_t)(i * 16) * CCH + k0 + 32, CCH);
#pragma unroll
        for (int t = 0; t < 4; ++t)
            b1[t] = load_frag(brow + (size_t)(t * 16) * CCH + k0 + 32, CCH);
        // consume buf0 @ k0
#pragma unroll
        for (int i = 0; i < 2; ++i)
#pragma unroll
            for (int t = 0; t < 4; ++t)
                acc[i][t] = wmma_bf16(a0[i], b0[t], acc[i][t]);
        // load buf0 @ k0+64
#pragma unroll
        for (int i = 0; i < 2; ++i)
            a0[i] = load_frag(arow + (size_t)(i * 16) * CCH + k0 + 64, CCH);
#pragma unroll
        for (int t = 0; t < 4; ++t)
            b0[t] = load_frag(brow + (size_t)(t * 16) * CCH + k0 + 64, CCH);
        // consume buf1 @ k0+32
#pragma unroll
        for (int i = 0; i < 2; ++i)
#pragma unroll
            for (int t = 0; t < 4; ++t)
                acc[i][t] = wmma_bf16(a1[i], b1[t], acc[i][t]);
    }
    // tail: buf0 holds k = CCH-64; load buf1 @ CCH-32
#pragma unroll
    for (int i = 0; i < 2; ++i)
        a1[i] = load_frag(arow + (size_t)(i * 16) * CCH + CCH - 32, CCH);
#pragma unroll
    for (int t = 0; t < 4; ++t)
        b1[t] = load_frag(brow + (size_t)(t * 16) * CCH + CCH - 32, CCH);
#pragma unroll
    for (int i = 0; i < 2; ++i)
#pragma unroll
        for (int t = 0; t < 4; ++t)
            acc[i][t] = wmma_bf16(a0[i], b0[t], acc[i][t]);
#pragma unroll
    for (int i = 0; i < 2; ++i)
#pragma unroll
        for (int t = 0; t < 4; ++t)
            acc[i][t] = wmma_bf16(a1[i], b1[t], acc[i][t]);

    const int hi = lane >> 4, nn = lane & 15;
#pragma unroll
    for (int i = 0; i < 2; ++i)
#pragma unroll
        for (int t = 0; t < 4; ++t)
#pragma unroll
            for (int r = 0; r < 8; ++r) {
                int m = mt * 32 + i * 16 + r + 8 * hi;
                int n = ng * 64 + t * 16 + nn;
                float v = acc[i][t][r] + bias[bias_on_m ? m : n];
                out[(size_t)m * Nn + n] = (bf16)v;
            }
}

// ---------------------------------------------------------------------------
// Flash attention: q[b][i][c], k[b][j][c], vT[b][c][j] (bf16) -> ha[b][i][c]
// One block = 16 query rows, 4 waves; wave w owns channel slice [128w,128w+128).
// ---------------------------------------------------------------------------
__global__ __launch_bounds__(128) void flash_kernel(const bf16* __restrict__ q,
                                                    const bf16* __restrict__ k,
                                                    const bf16* __restrict__ vT,
                                                    bf16* __restrict__ ha) {
    const int batch = blockIdx.y;
    const int i0    = blockIdx.x * 16;
    const int tid   = threadIdx.x;
    const int lane  = tid & 31;
    const int wave  = tid >> 5;
    const int hi    = lane >> 4;
    const int nn    = lane & 15;

    __shared__ float Sp[4][16][32];
    __shared__ float Sf[16][32];
    __shared__ bf16  Pt[16][32];
    __shared__ float mrow[16], lrow[16], crow[16];

    const bf16* qb = q  + ((size_t)batch * NPIX + i0) * CCH + wave * 128;
    const bf16* kb = k  +  (size_t)batch * NPIX * CCH        + wave * 128;
    const bf16* vb = vT +  (size_t)batch * CCH * NPIX + (size_t)wave * 128 * NPIX;

    v16bf qA[4];
#pragma unroll
    for (int t = 0; t < 4; ++t) qA[t] = load_frag(qb + t * 32, CCH);

    v8f O[8] = {};
    if (tid < 16) { mrow[tid] = -3.0e38f; lrow[tid] = 0.f; }
    __syncthreads();

    const float scale = 0.044194173824159216f;  // 512^-0.5

    for (int jt = 0; jt < NPIX / 32; ++jt) {
        const int j0 = jt * 32;

        // --- issue all K fragments (one clause), then all V fragments ---
        v16bf kf[8];
#pragma unroll
        for (int t = 0; t < 4; ++t) {
            kf[t]     = load_frag(kb + (size_t)j0 * CCH + t * 32, CCH);
            kf[4 + t] = load_frag(kb + (size_t)(j0 + 16) * CCH + t * 32, CCH);
        }
        v16bf vB[8];
#pragma unroll
        for (int ct = 0; ct < 8; ++ct)
            vB[ct] = load_frag(vb + (size_t)(ct * 16) * NPIX + j0, NPIX);

        // prefetch next j-tile K rows (global_prefetch_b8)
        if (jt + 1 < NPIX / 32) {
            const bf16* nk = kb + (size_t)(j0 + 32 + lane) * CCH;
            __builtin_prefetch(nk, 0, 1);
            __builtin_prefetch(nk + 64, 0, 1);
        }

        // --- partial S = Q_slice * K_slice^T (16 x 32, K=128) ---
        v8f s0 = {}, s1 = {};
#pragma unroll
        for (int t = 0; t < 4; ++t) {
            s0 = wmma_bf16(qA[t], kf[t],     s0);
            s1 = wmma_bf16(qA[t], kf[4 + t], s1);
        }
#pragma unroll
        for (int r = 0; r < 8; ++r) {
            int m = r + 8 * hi;
            Sp[wave][m][nn]      = s0[r];
            Sp[wave][m][nn + 16] = s1[r];
        }
        __syncthreads();
        // --- cross-wave reduce + scale ---
        {
            int e = tid * 4;
#pragma unroll
            for (int u = 0; u < 4; ++u) {
                int m = (e + u) >> 5, c = (e + u) & 31;
                Sf[m][c] = scale * (Sp[0][m][c] + Sp[1][m][c] + Sp[2][m][c] + Sp[3][m][c]);
            }
        }
        __syncthreads();
        // --- online softmax state update (one thread per row) ---
        if (tid < 16) {
            float mold = mrow[tid];
            float tmax = mold;
#pragma unroll
            for (int j = 0; j < 32; ++j) tmax = fmaxf(tmax, Sf[tid][j]);
            float corr = __expf(mold - tmax);
            float sum = 0.f;
#pragma unroll
            for (int j = 0; j < 32; ++j) {
                float p = __expf(Sf[tid][j] - tmax);
                Pt[tid][j] = (bf16)p;
                sum += p;
            }
            lrow[tid] = lrow[tid] * corr + sum;
            mrow[tid] = tmax;
            crow[tid] = corr;
        }
        __syncthreads();
        // --- rescale accumulators ---
#pragma unroll
        for (int r = 0; r < 8; ++r) {
            float c = crow[r + 8 * hi];
#pragma unroll
            for (int ct = 0; ct < 8; ++ct) O[ct][r] *= c;
        }
        // --- O += P * V_slice^T (V already in registers) ---
        v16bf pA = load_frag(&Pt[0][0], 32);   // from LDS (ds_load_b128)
#pragma unroll
        for (int ct = 0; ct < 8; ++ct)
            O[ct] = wmma_bf16(pA, vB[ct], O[ct]);
        __syncthreads();
    }

    // --- epilogue: divide by l, store bf16 [i][c] ---
#pragma unroll
    for (int ct = 0; ct < 8; ++ct)
#pragma unroll
        for (int r = 0; r < 8; ++r) {
            int m = r + 8 * hi;
            float v = O[ct][r] / lrow[m];
            ha[((size_t)batch * NPIX + i0 + m) * CCH + wave * 128 + ct * 16 + nn] = (bf16)v;
        }
}

// ---------------------------------------------------------------------------
// Proj + residual, ping-pong pipelined:
// out[b][o][n] = x[b][o][n] + sum_c ha[b][n][c]*pw[o][c] + pb[o]
// ---------------------------------------------------------------------------
__global__ __launch_bounds__(128) void proj_kernel(const bf16* __restrict__ A,   // ha
                                                   const bf16* __restrict__ Bt,  // p_w bf16
                                                   const float* __restrict__ pb,
                                                   const float* __restrict__ x,
                                                   float* __restrict__ out) {
    const int batch = blockIdx.z;
    A += (size_t)batch * NPIX * CCH;

    const int wave = threadIdx.x >> 5;
    const int lane = threadIdx.x & 31;
    const int wg   = blockIdx.x * 4 + wave;
    const int mt   = wg / 8;       // 128 m-tiles of 32 pixels
    const int ng   = wg % 8;       // 8 n-groups of 64 channels

    const bf16* arow = A  + (size_t)(mt * 32) * CCH;
    const bf16* brow = Bt + (size_t)(ng * 64) * CCH;

    v16bf a0[2], b0[4], a1[2], b1[4];
#pragma unroll
    for (int i = 0; i < 2; ++i) a0[i] = load_frag(arow + (size_t)(i * 16) * CCH, CCH);
#pragma unroll
    for (int t = 0; t < 4; ++t) b0[t] = load_frag(brow + (size_t)(t * 16) * CCH, CCH);

    v8f acc[2][4] = {};
    for (int k0 = 0; k0 < CCH - 64; k0 += 64) {
#pragma unroll
        for (int i = 0; i < 2; ++i)
            a1[i] = load_frag(arow + (size_t)(i * 16) * CCH + k0 + 32, CCH);
#pragma unroll
        for (int t = 0; t < 4; ++t)
            b1[t] = load_frag(brow + (size_t)(t * 16) * CCH + k0 + 32, CCH);
#pragma unroll
        for (int i = 0; i < 2; ++i)
#pragma unroll
            for (int t = 0; t < 4; ++t)
                acc[i][t] = wmma_bf16(a0[i], b0[t], acc[i][t]);
#pragma unroll
        for (int i = 0; i < 2; ++i)
            a0[i] = load_frag(arow + (size_t)(i * 16) * CCH + k0 + 64, CCH);
#pragma unroll
        for (int t = 0; t < 4; ++t)
            b0[t] = load_frag(brow + (size_t)(t * 16) * CCH + k0 + 64, CCH);
#pragma unroll
        for (int i = 0; i < 2; ++i)
#pragma unroll
            for (int t = 0; t < 4; ++t)
                acc[i][t] = wmma_bf16(a1[i], b1[t], acc[i][t]);
    }
#pragma unroll
    for (int i = 0; i < 2; ++i)
        a1[i] = load_frag(arow + (size_t)(i * 16) * CCH + CCH - 32, CCH);
#pragma unroll
    for (int t = 0; t < 4; ++t)
        b1[t] = load_frag(brow + (size_t)(t * 16) * CCH + CCH - 32, CCH);
#pragma unroll
    for (int i = 0; i < 2; ++i)
#pragma unroll
        for (int t = 0; t < 4; ++t)
            acc[i][t] = wmma_bf16(a0[i], b0[t], acc[i][t]);
#pragma unroll
    for (int i = 0; i < 2; ++i)
#pragma unroll
        for (int t = 0; t < 4; ++t)
            acc[i][t] = wmma_bf16(a1[i], b1[t], acc[i][t]);

    const int hi = lane >> 4, nn = lane & 15;
#pragma unroll
    for (int i = 0; i < 2; ++i)
#pragma unroll
        for (int t = 0; t < 4; ++t)
#pragma unroll
            for (int r = 0; r < 8; ++r) {
                int m = mt * 32 + i * 16 + r + 8 * hi;            // pixel
                int n = ng * 64 + t * 16 + nn;                    // channel
                size_t idx = ((size_t)batch * CCH + n) * NPIX + m;
                out[idx] = x[idx] + acc[i][t][r] + pb[n];
            }
}

// ---------------------------------------------------------------------------
extern "C" void kernel_launch(void* const* d_in, const int* in_sizes, int n_in,
                              void* d_out, int out_size, void* d_ws, size_t ws_size,
                              hipStream_t stream) {
    const float* x    = (const float*)d_in[0];
    const float* gn_w = (const float*)d_in[1];
    const float* gn_b = (const float*)d_in[2];
    const float* q_w  = (const float*)d_in[3];
    const float* q_b  = (const float*)d_in[4];
    const float* k_w  = (const float*)d_in[5];
    const float* k_b  = (const float*)d_in[6];
    const float* v_w  = (const float*)d_in[7];
    const float* v_b  = (const float*)d_in[8];
    const float* p_w  = (const float*)d_in[9];
    const float* p_b  = (const float*)d_in[10];

    char* ws = (char*)d_ws;
    const size_t NC   = (size_t)BATCH * NPIX * CCH;
    const size_t SZNC = NC * sizeof(bf16);
    const size_t SZW  = (size_t)CCH * CCH * sizeof(bf16);

    bf16* hn = (bf16*)(ws);
    bf16* qb = (bf16*)(ws + 1 * SZNC);
    bf16* kb = (bf16*)(ws + 2 * SZNC);
    bf16* vT = (bf16*)(ws + 3 * SZNC);
    bf16* ha = (bf16*)(ws + 4 * SZNC);
    bf16* wq = (bf16*)(ws + 5 * SZNC);
    bf16* wk = (bf16*)(ws + 5 * SZNC + 1 * SZW);
    bf16* wv = (bf16*)(ws + 5 * SZNC + 2 * SZW);
    bf16* wp = (bf16*)(ws + 5 * SZNC + 3 * SZW);

    const int nW = CCH * CCH;
    cvt_bf16_kernel<<<nW / 256, 256, 0, stream>>>(q_w, wq, nW);
    cvt_bf16_kernel<<<nW / 256, 256, 0, stream>>>(k_w, wk, nW);
    cvt_bf16_kernel<<<nW / 256, 256, 0, stream>>>(v_w, wv, nW);
    cvt_bf16_kernel<<<nW / 256, 256, 0, stream>>>(p_w, wp, nW);

    gn_kernel<<<BATCH * 32, 256, 0, stream>>>(x, gn_w, gn_b, hn);

    const long long bNC = (long long)NPIX * CCH;
    // q[n][o], k[n][o]: A=hn [4096x512], Bt=W [512x512]; 1024 waves/batch
    gemm_nt_kernel<<<dim3(256, 1, BATCH), 128, 0, stream>>>(
        hn, bNC, wq, 0, q_b, 0, qb, bNC, NPIX, CCH, 8);
    gemm_nt_kernel<<<dim3(256, 1, BATCH), 128, 0, stream>>>(
        hn, bNC, wk, 0, k_b, 0, kb, bNC, NPIX, CCH, 8);
    // vT[o][n]: A=Wv [512x512], Bt=hn [4096x512]
    gemm_nt_kernel<<<dim3(256, 1, BATCH), 128, 0, stream>>>(
        wv, 0, hn, bNC, v_b, 1, vT, bNC, CCH, NPIX, 64);

    flash_kernel<<<dim3(NPIX / 16, BATCH), 128, 0, stream>>>(qb, kb, vT, ha);

    proj_kernel<<<dim3(256, 1, BATCH), 128, 0, stream>>>(ha, wp, p_b, x, (float*)d_out);
}
